// BahdanauAttention_84636625535499
// MI455X (gfx1250) — compile-verified, matrix operand order checked by reading, not run on hardware
//
#include <hip/hip_runtime.h>
#include <hip/hip_bf16.h>
#include <stdint.h>

#define B_ 32
#define S_ 2048
#define H_ 1024
#define M_ (B_ * S_)   // 65536 rows, m = s*B + b

typedef __attribute__((ext_vector_type(16))) __bf16        v16bf;
typedef __attribute__((ext_vector_type(8)))  float         v8f;
typedef __attribute__((ext_vector_type(4)))  unsigned int  v4u;

__device__ __forceinline__ unsigned int pack_bf16x2(float lo, float hi) {
    unsigned int a = __float_as_uint(lo);
    unsigned int b = __float_as_uint(hi);
    a = (a + 0x7FFFu + ((a >> 16) & 1u)) >> 16;   // RNE f32 -> bf16
    b = (b + 0x7FFFu + ((b >> 16) & 1u)) >> 16;
    return a | (b << 16);
}

// Fast hardware tanh: V_TANH_F32 is a native TRANS op on gfx1250.
__device__ __forceinline__ float fast_tanh(float x) {
#if defined(__has_builtin) && __has_builtin(__builtin_amdgcn_tanhf)
    return __builtin_amdgcn_tanhf(x);
#else
    float y;
    asm volatile("v_tanh_f32 %0, %1" : "=v"(y) : "v"(x));
    return y;
#endif
}

// ---------------------------------------------------------------- zero ctx
__global__ void __launch_bounds__(256) zero_kernel(float* p, int n) {
    int i = blockIdx.x * 256 + threadIdx.x;
    if (i < n) p[i] = 0.f;
}

// ------------------------------------------- hidProj[b,n] = hid.W_h^T + b
__global__ void __launch_bounds__(256)
hidproj_kernel(const float* __restrict__ hidden, const float* __restrict__ attn_W,
               const float* __restrict__ attn_b, float* __restrict__ hidProj) {
    int b = blockIdx.x >> 2;                          // 32
    int n = ((blockIdx.x & 3) << 8) + threadIdx.x;    // 1024
    const float* hrow = hidden + b * H_;
    const float* wrow = attn_W + (size_t)n * (2 * H_);  // hid half: k in [0,H)
    float acc = 0.f;
    for (int k = 0; k < H_; ++k) acc += hrow[k] * wrow[k];
    hidProj[b * H_ + n] = acc + attn_b[n];
}

// ---------------- W_e (enc half of attn_W) -> bf16, WMMA-B fragment order
// layout: [ntile(64)][kchunk(32)][lane(32)][8 dwords (16 bf16)]
__global__ void __launch_bounds__(256)
wswz_kernel(const float* __restrict__ attn_W, unsigned int* __restrict__ wsw) {
    int idx = blockIdx.x * 256 + threadIdx.x;   // 0 .. 1024*512-1
    int n  = idx >> 9;
    int k  = (idx & 511) << 1;
    const float* src = attn_W + (size_t)n * (2 * H_) + H_ + k;
    unsigned int pk = pack_bf16x2(src[0], src[1]);
    int kin  = k & 31, c = k >> 5;
    int hi   = (kin >> 3) & 1;                       // K {8..15,24..31} -> lanes 16..31
    int lane = (n & 15) + (hi << 4);
    int p    = ((kin >> 4) << 2) + ((kin & 7) >> 1); // dword slot within lane
    wsw[((((n >> 4) * 32 + c) * 32 + lane) << 3) + p] = pk;
}

// ------- main fused kernel: D = enc x W_e^T + hidProj ; score = v . tanh(D)
__global__ void __launch_bounds__(256)
energy_score_kernel(const float* __restrict__ enc,        // [M_, H_]
                    const unsigned int* __restrict__ wsw, // swizzled bf16 W_e
                    const float* __restrict__ hidProj,    // [B_, H_]
                    const float* __restrict__ vW,         // [H_]
                    float* __restrict__ scores) {         // [M_]
    extern __shared__ unsigned int lds[];   // 8 waves * 8192 dwords = 256 KB
    const int t  = threadIdx.x;
    const int m0 = blockIdx.x * 128;

    // ---- stage A tile (128 rows x 1024 K) as bf16, A-fragment swizzled ----
    for (int i = 0; i < 256; ++i) {
        int idx = i * 256 + t;
        int row = idx >> 9;            // 512 k-pairs per row
        int k   = (idx & 511) << 1;
        float2 v = *(const float2*)(enc + (size_t)(m0 + row) * H_ + k);
        unsigned int pk = pack_bf16x2(v.x, v.y);
        int kin  = k & 31, c = k >> 5;
        int hi   = (kin >> 3) & 1;
        int lane = (row & 15) + (hi << 4);
        int p    = ((kin >> 4) << 2) + ((kin & 7) >> 1);
        lds[((row >> 4) << 13) + (c << 8) + (lane << 3) + p] = pk;
    }
    __syncthreads();

    const int w   = t >> 5;          // wave 0..7, owns rows m0 + w*16 ..
    const int l   = t & 31;
    const int col = l & 15;          // D column within n-tile
    const int hi  = l >> 4;          // D rows: g + 8*hi
    const unsigned int* aw = lds + (w << 13) + (l << 3);
    const int brow = ((w & 1) << 4) + (hi << 3);   // b = brow + g (m0 % 32 == 0)

    float srow[8];
#pragma unroll
    for (int g = 0; g < 8; ++g) srow[g] = 0.f;

    for (int nt = 0; nt < 64; ++nt) {
        const int n0 = nt << 4;
        const float vv = vW[n0 + col];
        v8f c;
        const float* hp = hidProj + (size_t)brow * H_ + n0 + col;
#pragma unroll
        for (int g = 0; g < 8; ++g) c[g] = hp[(size_t)g * H_];   // bias in C

        const unsigned int* bbase = wsw + ((size_t)nt << 13);
#pragma unroll 4
        for (int kc = 0; kc < 32; ++kc) {
            union { v16bf bf; v4u u[2]; } A, Bm;
            const v4u* ap = (const v4u*)(aw + (kc << 8));
            A.u[0] = ap[0]; A.u[1] = ap[1];
            const v4u* bp = (const v4u*)(bbase + (((kc << 5) + l) << 3));
            Bm.u[0] = bp[0]; Bm.u[1] = bp[1];
            c = __builtin_amdgcn_wmma_f32_16x16x32_bf16(
                    false, A.bf, false, Bm.bf, (short)0, c, false, false);
        }
#pragma unroll
        for (int g = 0; g < 8; ++g) srow[g] += fast_tanh(c[g]) * vv;
    }
    // reduce over the 16 columns (lanes within each 16-lane half)
#pragma unroll
    for (int g = 0; g < 8; ++g) {
        float x = srow[g];
        x += __shfl_xor(x, 1, 32);
        x += __shfl_xor(x, 2, 32);
        x += __shfl_xor(x, 4, 32);
        x += __shfl_xor(x, 8, 32);
        srow[g] = x;
    }
    if (col == 0) {
        int mbase = m0 + (w << 4) + (hi << 3);
#pragma unroll
        for (int g = 0; g < 8; ++g) scores[mbase + g] = srow[g];
    }
}

// ------------------------------------------------ softmax over s (per b)
__global__ void __launch_bounds__(256)
softmax_kernel(const float* __restrict__ scores, float* __restrict__ outw) {
    __shared__ float red[256];
    const int b = blockIdx.x, t = threadIdx.x;
    float mx = -3.4e38f;
    for (int s = t; s < S_; s += 256) mx = fmaxf(mx, scores[s * B_ + b]);
    red[t] = mx; __syncthreads();
    for (int o = 128; o > 0; o >>= 1) { if (t < o) red[t] = fmaxf(red[t], red[t + o]); __syncthreads(); }
    mx = red[0]; __syncthreads();
    float sum = 0.f;
    for (int s = t; s < S_; s += 256) sum += __expf(scores[s * B_ + b] - mx);
    red[t] = sum; __syncthreads();
    for (int o = 128; o > 0; o >>= 1) { if (t < o) red[t] += red[t + o]; __syncthreads(); }
    const float inv = 1.0f / red[0];
    for (int s = t; s < S_; s += 256)
        outw[b * S_ + s] = __expf(scores[s * B_ + b] - mx) * inv;
}

// ----------------------------------- context[b,h] = sum_s w[b,s]*enc[b,s,h]
__global__ void __launch_bounds__(256)
context_kernel(const float* __restrict__ enc, const float* __restrict__ attnw,
               float* __restrict__ ctx) {
    const int b  = blockIdx.z;
    const int h  = blockIdx.y * 256 + threadIdx.x;
    const int s0 = blockIdx.x * 256;
    const float* wp = attnw + b * S_ + s0;
    const float* ep = enc + ((size_t)s0 * B_ + b) * H_ + h;
    float acc = 0.f;
    for (int s = 0; s < 256; ++s) acc += wp[s] * ep[(size_t)s * B_ * H_];
    unsafeAtomicAdd(&ctx[b * H_ + h], acc);
}

extern "C" void kernel_launch(void* const* d_in, const int* in_sizes, int n_in,
                              void* d_out, int out_size, void* d_ws, size_t ws_size,
                              hipStream_t stream) {
    const float* hidden = (const float*)d_in[0];
    const float* enc    = (const float*)d_in[1];   // [S,B,H] == [M_, H_] row-major
    const float* attn_W = (const float*)d_in[2];   // [H, 2H]
    const float* attn_b = (const float*)d_in[3];   // [H]
    const float* v_W    = (const float*)d_in[4];   // [1, H]

    float* ctx   = (float*)d_out;                  // [B, H]
    float* attnw = (float*)d_out + B_ * H_;        // [B, S]

    unsigned int* wsw = (unsigned int*)d_ws;                                   // 2 MB
    float* hidProj = (float*)((char*)d_ws + 2u * 1024u * 1024u);               // 128 KB
    float* scores  = (float*)((char*)d_ws + 2u * 1024u * 1024u + 131072u);     // 256 KB

    zero_kernel<<<(B_ * H_ + 255) / 256, 256, 0, stream>>>(ctx, B_ * H_);
    hidproj_kernel<<<128, 256, 0, stream>>>(hidden, attn_W, attn_b, hidProj);
    wswz_kernel<<<2048, 256, 0, stream>>>(attn_W, wsw);
    energy_score_kernel<<<M_ / 128, 256, 262144, stream>>>(enc, wsw, hidProj, v_W, scores);
    softmax_kernel<<<B_, 256, 0, stream>>>(scores, attnw);
    context_kernel<<<dim3(8, 4, 32), 256, 0, stream>>>(enc, attnw, ctx);
}
